// SoftmaxAttention_61778809586102
// MI455X (gfx1250) — matrix operand order repaired and checked by reading, not compiled, not run on hardware
//
#include <hip/hip_runtime.h>
#include <hip/hip_bf16.h>

typedef __bf16 bf16_t;
typedef __attribute__((ext_vector_type(16))) __bf16 v16bf;
typedef __attribute__((ext_vector_type(8)))  __bf16 v8bf;
typedef __attribute__((ext_vector_type(8)))  float  v8f;

#define B_  4
#define N_  2048
#define E_  1024
#define H_  16
#define D_  64

// ---- CDNA5 async global->LDS copy (ASYNCcnt-tracked, CDNA5 ISA) ----------
#define ASYNC_B128(ldsoff, gptr)                                              \
  asm volatile("global_load_async_to_lds_b128 %0, %1, off"                    \
               :: "v"(ldsoff), "v"(gptr) : "memory")

#if __has_builtin(__builtin_amdgcn_s_wait_asynccnt)
#define WAIT_ASYNC(n) __builtin_amdgcn_s_wait_asynccnt(n)
#else
#define WAIT_ASYNC(n) asm volatile("s_wait_asynccnt " #n ::: "memory")
#endif

static __device__ __forceinline__ unsigned lds_off(const void* p) {
  return (unsigned)(unsigned long long)p;  // low 32 bits of flat addr = LDS offset
}

static __device__ __forceinline__ v16bf make_afrag(v8bf lo, v8bf hi) {
  return __builtin_shufflevector(lo, hi, 0,1,2,3,4,5,6,7,8,9,10,11,12,13,14,15);
}

static __device__ __forceinline__ v8f wmma_bf16(v16bf a, v16bf b, v8f c) {
  return __builtin_amdgcn_wmma_f32_16x16x32_bf16(false, a, false, b, (short)0, c,
                                                 false, false);
}

// ---------------------------------------------------------------- fp32 -> bf16
__global__ void cvt_bf16_kernel(const float* __restrict__ in,
                                bf16_t* __restrict__ out, int n) {
  int i = blockIdx.x * blockDim.x + threadIdx.x;
  int stride = gridDim.x * blockDim.x;
  for (; i < n; i += stride) out[i] = (bf16_t)in[i];
}

// ------------------------------------------------- QKV projection (bf16 WMMA)
// Y[m,f] = sum_k X[m,k]*W[f,k]; M=8192, K=1024, F=3072.
// Block (8 waves) computes 128(M) x 64(F); weight tile 64x32 staged in LDS by
// async loads, double buffered. Q gets 1/sqrt(D) folded in; V stored [BH,D,N].
__global__ void gemm_qkv_kernel(const bf16_t* __restrict__ X,
                                const bf16_t* __restrict__ W,
                                bf16_t* __restrict__ Q,
                                bf16_t* __restrict__ K,
                                bf16_t* __restrict__ V) {
  __shared__ __align__(32) bf16_t wtile[2 * 64 * 32];  // 8 KB, double buffered
  const int NT = 48;
  int mblk = blockIdx.x / NT, ntile = blockIdx.x % NT;
  int tid  = threadIdx.x;
  int w    = tid >> 5, lane = tid & 31;
  int hi   = lane >> 4, ln = lane & 15;

  // per-thread async staging slot: 256 threads x 16B = 4KB tile
  int brow = tid >> 2, bseg = tid & 3;  // row 0..63, 16B segment 0..3
  const bf16_t* wsrc = W + (ntile * 64 + brow) * E_ + bseg * 8;
  unsigned ldst = lds_off(wtile) + brow * 64 + bseg * 16;

  ASYNC_B128(ldst, wsrc);  // prefetch kb=0 into buffer 0

  const bf16_t* xrow = X + (mblk * 128 + w * 16 + ln) * E_;
  v8f acc0 = {}, acc1 = {}, acc2 = {}, acc3 = {};

  for (int kb = 0; kb < E_; kb += 32) {
    unsigned cur = (kb >> 5) & 1;
    v16bf a = *(const v16bf*)(xrow + kb + 16 * hi);
    if (kb + 32 < E_) {
      ASYNC_B128(ldst + (cur ^ 1) * 4096, wsrc + kb + 32);
      WAIT_ASYNC(1);                       // previous tile landed
    } else {
      WAIT_ASYNC(0);
    }
    __syncthreads();                       // tile visible to all waves

    // batch all LDS fragment loads, then fire the WMMAs back-to-back
    const char* bt = (const char*)wtile + cur * 4096;
    v16bf b0 = *(const v16bf*)(bt + (0 * 16 + ln) * 64 + 32 * hi);
    v16bf b1 = *(const v16bf*)(bt + (1 * 16 + ln) * 64 + 32 * hi);
    v16bf b2 = *(const v16bf*)(bt + (2 * 16 + ln) * 64 + 32 * hi);
    v16bf b3 = *(const v16bf*)(bt + (3 * 16 + ln) * 64 + 32 * hi);
    acc0 = wmma_bf16(a, b0, acc0);
    acc1 = wmma_bf16(a, b1, acc1);
    acc2 = wmma_bf16(a, b2, acc2);
    acc3 = wmma_bf16(a, b3, acc3);
    __syncthreads();                       // done reading before next overwrite
  }

  v8f accs[4] = {acc0, acc1, acc2, acc3};
#pragma unroll
  for (int j = 0; j < 4; ++j) {
    int f = ntile * 64 + j * 16 + ln;
    int sec = f >> 10;  // 0=Q 1=K 2=V
    int e = f & (E_ - 1);
    int h = e >> 6, d = e & (D_ - 1);
#pragma unroll
    for (int i = 0; i < 8; ++i) {
      int tok = mblk * 128 + w * 16 + i + 8 * hi;
      int bb = tok >> 11, n = tok & (N_ - 1);
      int bh = bb * H_ + h;
      float v = accs[j][i];
      if (sec == 0)      Q[(bh * N_ + n) * D_ + d] = (bf16_t)(v * 0.125f);
      else if (sec == 1) K[(bh * N_ + n) * D_ + d] = (bf16_t)v;
      else               V[(bh * D_ + d) * N_ + n] = (bf16_t)v;
    }
  }
}

// --------------------------------------------------- flash attention
// Block = 4 waves sharing one (b,h); wave w owns query tile qt.
// Per key chunk (32 keys): K tile 32x64 (4KB) and V tile 64x32 (4KB) staged
// into double-buffered LDS with async loads, shared by all 4 waves.
// Per-wave scratch: f32 scores 16x32 | bf16 P 16x32 | alpha[16] | sum[16].
__global__ void attention_kernel(const bf16_t* __restrict__ Q,
                                 const bf16_t* __restrict__ Kb,
                                 const bf16_t* __restrict__ V,
                                 bf16_t* __restrict__ ctx) {
  __shared__ __align__(32) char smem[2 * 4096 + 2 * 4096 + 4 * 3200];
  char* kt_base = smem;                 // [2][32 rows of 128B] = 8KB
  char* vt_base = smem + 8192;          // [2][64 rows of 64B]  = 8KB
  int tid  = threadIdx.x;
  int wave = tid >> 5, lane = tid & 31;
  int hi   = lane >> 4, ln = lane & 15;
  int bh   = blockIdx.x >> 5;
  int qt   = (blockIdx.x & 31) * 4 + wave;
  int qbase = qt * 16;

  char*   scratch = smem + 16384 + wave * 3200;
  float*  sc   = (float*)scratch;            // 2048 B
  bf16_t* p    = (bf16_t*)(scratch + 2048);  // 1024 B
  float*  arow = (float*)(scratch + 3072);   // 64 B
  float*  srow = (float*)(scratch + 3136);   // 64 B

  // async staging: 128 threads, 2 x b128 per tile per thread
  // K tile: 32 rows x 128B ; V tile: 64 rows x 64B
  int id0 = tid, id1 = tid + 128;
  const bf16_t* ksrc0 = Kb + (bh * N_ + (id0 >> 3)) * D_ + (id0 & 7) * 8;
  const bf16_t* ksrc1 = Kb + (bh * N_ + (id1 >> 3)) * D_ + (id1 & 7) * 8;
  unsigned kd0 = lds_off(kt_base) + (id0 >> 3) * 128 + (id0 & 7) * 16;
  unsigned kd1 = lds_off(kt_base) + (id1 >> 3) * 128 + (id1 & 7) * 16;
  const bf16_t* vsrc0 = V + (bh * D_ + (id0 >> 2)) * N_ + (id0 & 3) * 8;
  const bf16_t* vsrc1 = V + (bh * D_ + (id1 >> 2)) * N_ + (id1 & 3) * 8;
  unsigned vd0 = lds_off(vt_base) + (id0 >> 2) * 64 + (id0 & 3) * 16;
  unsigned vd1 = lds_off(vt_base) + (id1 >> 2) * 64 + (id1 & 3) * 16;

  // prefetch chunk 0 into buffer 0
  ASYNC_B128(kd0, ksrc0);
  ASYNC_B128(kd1, ksrc1);
  ASYNC_B128(vd0, vsrc0);
  ASYNC_B128(vd1, vsrc1);

  const bf16_t* qrow = Q + (bh * N_ + qbase + ln) * D_;
  v16bf a0 = *(const v16bf*)(qrow + 16 * hi);
  v16bf a1 = *(const v16bf*)(qrow + 32 + 16 * hi);

  v8f o0 = {}, o1 = {}, o2 = {}, o3 = {};
  float m_r = -3.0e38f, s_r = 0.0f;

  for (int kc = 0; kc < N_; kc += 32) {
    unsigned cur = (kc >> 5) & 1;
    if (kc + 32 < N_) {  // prefetch next chunk (keys are [BH,N,D]; Vt [BH,D,N])
      unsigned boff = (cur ^ 1) * 4096;
      ASYNC_B128(kd0 + boff, ksrc0 + (kc + 32) * D_);
      ASYNC_B128(kd1 + boff, ksrc1 + (kc + 32) * D_);
      ASYNC_B128(vd0 + boff, vsrc0 + (kc + 32));
      ASYNC_B128(vd1 + boff, vsrc1 + (kc + 32));
      WAIT_ASYNC(4);                     // current chunk's 4 ops complete
    } else {
      WAIT_ASYNC(0);
    }
    __syncthreads();                     // tiles visible to all 4 waves

    // scores: 16 queries x 32 keys, two WMMA k-steps over D=64
    const char* kt = kt_base + cur * 4096;
    v16bf k00 = *(const v16bf*)(kt + ln * 128 + 32 * hi);
    v16bf k01 = *(const v16bf*)(kt + ln * 128 + 64 + 32 * hi);
    v16bf k10 = *(const v16bf*)(kt + (16 + ln) * 128 + 32 * hi);
    v16bf k11 = *(const v16bf*)(kt + (16 + ln) * 128 + 64 + 32 * hi);
    v8f s0 = {}, s1 = {};
    s0 = wmma_bf16(a0, k00, s0);
    s0 = wmma_bf16(a1, k01, s0);
    s1 = wmma_bf16(a0, k10, s1);
    s1 = wmma_bf16(a1, k11, s1);

#pragma unroll
    for (int i = 0; i < 8; ++i) {
      sc[(i + 8 * hi) * 32 + ln]      = s0[i];
      sc[(i + 8 * hi) * 32 + 16 + ln] = s1[i];
    }
    __syncthreads();

    // online softmax: one query row per lane (lanes 0..15)
    if (lane < 16) {
      const float* scr = sc + lane * 32;
      float mx = scr[0];
#pragma unroll
      for (int c = 1; c < 32; ++c) mx = fmaxf(mx, scr[c]);
      float mnew  = fmaxf(m_r, mx);
      float alpha = __expf(m_r - mnew);
      float sum = 0.f;
#pragma unroll
      for (int c = 0; c < 32; ++c) {
        float ev = __expf(scr[c] - mnew);
        p[lane * 32 + c] = (bf16_t)ev;
        sum += ev;
      }
      s_r = s_r * alpha + sum;
      m_r = mnew;
      arow[lane] = alpha;
    }
    __syncthreads();

#pragma unroll
    for (int i = 0; i < 8; ++i) {
      float f = arow[i + 8 * hi];
      o0[i] *= f; o1[i] *= f; o2[i] *= f; o3[i] *= f;
    }

    // P fragment + all V fragments first, then back-to-back WMMAs
    v16bf pa = *(const v16bf*)(p + ln * 32 + 16 * hi);
    const char* vt = vt_base + cur * 4096;
    v16bf vb0 = *(const v16bf*)(vt + (0 * 16 + ln) * 64 + 32 * hi);
    v16bf vb1 = *(const v16bf*)(vt + (1 * 16 + ln) * 64 + 32 * hi);
    v16bf vb2 = *(const v16bf*)(vt + (2 * 16 + ln) * 64 + 32 * hi);
    v16bf vb3 = *(const v16bf*)(vt + (3 * 16 + ln) * 64 + 32 * hi);
    o0 = wmma_bf16(pa, vb0, o0);
    o1 = wmma_bf16(pa, vb1, o1);
    o2 = wmma_bf16(pa, vb2, o2);
    o3 = wmma_bf16(pa, vb3, o3);
    __syncthreads();                     // done with tiles + scratch
  }

  if (lane < 16) srow[lane] = s_r;
  __syncthreads();

  int b = bh / H_, h = bh % H_;
#pragma unroll
  for (int i = 0; i < 8; ++i) {
    int n = qbase + i + 8 * hi;
    float inv = 1.0f / srow[i + 8 * hi];
    bf16_t* crow = ctx + (b * N_ + n) * E_ + h * D_;
    crow[0 * 16 + ln] = (bf16_t)(o0[i] * inv);
    crow[1 * 16 + ln] = (bf16_t)(o1[i] * inv);
    crow[2 * 16 + ln] = (bf16_t)(o2[i] * inv);
    crow[3 * 16 + ln] = (bf16_t)(o3[i] * inv);
  }
}

// ------------------------------------------------- output projection -> fp32
__global__ void gemm_out_kernel(const bf16_t* __restrict__ X,   // ctx [8192,1024]
                                const bf16_t* __restrict__ W,   // [1024,1024]
                                float* __restrict__ out) {
  __shared__ __align__(32) bf16_t wtile[2 * 64 * 32];
  const int NT = 16;
  int mblk = blockIdx.x / NT, ntile = blockIdx.x % NT;
  int tid  = threadIdx.x;
  int w    = tid >> 5, lane = tid & 31;
  int hi   = lane >> 4, ln = lane & 15;

  int brow = tid >> 2, bseg = tid & 3;
  const bf16_t* wsrc = W + (ntile * 64 + brow) * E_ + bseg * 8;
  unsigned ldst = lds_off(wtile) + brow * 64 + bseg * 16;

  ASYNC_B128(ldst, wsrc);

  const bf16_t* xrow = X + (mblk * 128 + w * 16 + ln) * E_;
  v8f acc0 = {}, acc1 = {}, acc2 = {}, acc3 = {};

  for (int kb = 0; kb < E_; kb += 32) {
    unsigned cur = (kb >> 5) & 1;
    v16bf a = *(const v16bf*)(xrow + kb + 16 * hi);
    if (kb + 32 < E_) {
      ASYNC_B128(ldst + (cur ^ 1) * 4096, wsrc + kb + 32);
      WAIT_ASYNC(1);
    } else {
      WAIT_ASYNC(0);
    }
    __syncthreads();
    const char* bt = (const char*)wtile + cur * 4096;
    v16bf b0 = *(const v16bf*)(bt + (0 * 16 + ln) * 64 + 32 * hi);
    v16bf b1 = *(const v16bf*)(bt + (1 * 16 + ln) * 64 + 32 * hi);
    v16bf b2 = *(const v16bf*)(bt + (2 * 16 + ln) * 64 + 32 * hi);
    v16bf b3 = *(const v16bf*)(bt + (3 * 16 + ln) * 64 + 32 * hi);
    acc0 = wmma_bf16(a, b0, acc0);
    acc1 = wmma_bf16(a, b1, acc1);
    acc2 = wmma_bf16(a, b2, acc2);
    acc3 = wmma_bf16(a, b3, acc3);
    __syncthreads();
  }

  v8f accs[4] = {acc0, acc1, acc2, acc3};
#pragma unroll
  for (int j = 0; j < 4; ++j) {
    int f = ntile * 64 + j * 16 + ln;
#pragma unroll
    for (int i = 0; i < 8; ++i) {
      int tok = mblk * 128 + w * 16 + i + 8 * hi;
      out[tok * E_ + f] = accs[j][i];
    }
  }
}

extern "C" void kernel_launch(void* const* d_in, const int* in_sizes, int n_in,
                              void* d_out, int out_size, void* d_ws, size_t ws_size,
                              hipStream_t stream) {
  const float* x     = (const float*)d_in[0];  // [B,N,E]
  const float* w_qkv = (const float*)d_in[1];  // [3E,E]
  const float* w_out = (const float*)d_in[2];  // [E,E]
  float* out = (float*)d_out;

  const size_t XE = (size_t)B_ * N_ * E_;      // 8,388,608
  bf16_t* xb    = (bf16_t*)d_ws;
  bf16_t* wqkvb = xb + XE;
  bf16_t* woutb = wqkvb + (size_t)3 * E_ * E_;
  bf16_t* qb    = woutb + (size_t)E_ * E_;
  bf16_t* kb    = qb + XE;
  bf16_t* vb    = kb + XE;
  bf16_t* ctx   = vb + XE;

  cvt_bf16_kernel<<<2048, 256, 0, stream>>>(x, xb, (int)XE);
  cvt_bf16_kernel<<<1024, 256, 0, stream>>>(w_qkv, wqkvb, 3 * E_ * E_);
  cvt_bf16_kernel<<<512, 256, 0, stream>>>(w_out, woutb, E_ * E_);

  // 64 M-blocks * 48 F-tiles
  gemm_qkv_kernel<<<3072, 256, 0, stream>>>(xb, wqkvb, qb, kb, vb);

  // B*H = 64 bh * 32 blocks (4 query tiles each)
  attention_kernel<<<2048, 128, 0, stream>>>(qb, kb, vb, ctx);

  // 64 M-blocks * 16 F-tiles
  gemm_out_kernel<<<1024, 256, 0, stream>>>(ctx, woutb, out);
}
